// MambaBackbone_60490319396873
// MI455X (gfx1250) — compile-verified, matrix operand order checked by reading, not run on hardware
//
#include <hip/hip_runtime.h>
#include <hip/hip_bf16.h>
#include <math.h>

// ---------------- problem constants ----------------
#define BB   2
#define LL   1024
#define DD   768
#define NLAY 8
#define DI   1536           // 2*D
#define NS   16
#define DR   48             // D/16
#define KC   4
#define EPSV 1e-5f
#define ROWS (BB*LL)        // 2048

typedef __attribute__((ext_vector_type(16))) _Float16 v16h;
typedef __attribute__((ext_vector_type(8)))  _Float16 h8;
typedef __attribute__((ext_vector_type(8)))  float    v8f;
typedef __attribute__((ext_vector_type(4)))  int      i4;

#define AS1 __attribute__((address_space(1)))
#define AS3 __attribute__((address_space(3)))
typedef AS1 i4 gi4;   // int4 in global address space
typedef AS3 i4 li4;   // int4 in LDS address space

#if defined(__has_builtin)
#if __has_builtin(__builtin_amdgcn_global_load_async_to_lds_b128)
#define ASYNC_COPY 1
#endif
#endif

static __device__ __forceinline__ void async_wait0() {
#if defined(ASYNC_COPY)
#if __has_builtin(__builtin_amdgcn_s_wait_asynccnt)
    __builtin_amdgcn_s_wait_asynccnt(0);
#else
    asm volatile("s_wait_asynccnt 0x0" ::: "memory");
#endif
#endif
}

static __device__ __forceinline__ void async_b128(const _Float16* g,
                                                  _Float16* l) {
#if defined(ASYNC_COPY)
    __builtin_amdgcn_global_load_async_to_lds_b128(
        (gi4*)(AS1 const void*)g, (li4*)(AS3 void*)l, 0, 0);
#else
    *(h8*)l = *(const h8*)g;
#endif
}

// ============================================================
// Block-wide sum reduction (256 threads)
// ============================================================
__device__ __forceinline__ float block_sum256(float v, float* sbuf) {
    int tid = threadIdx.x;
    sbuf[tid] = v;
    __syncthreads();
    #pragma unroll
    for (int s = 128; s > 0; s >>= 1) {
        if (tid < s) sbuf[tid] += sbuf[tid + s];
        __syncthreads();
    }
    float r = sbuf[0];
    __syncthreads();
    return r;
}

// ============================================================
// Residual accumulate + LayerNorm. One block per row (D=768).
//   res_out = first ? x : x + res_in      (in-place safe)
//   xn_h    = f16( LN(x)*w + b )          (GEMM operand, f16 only)
// ============================================================
__global__ __launch_bounds__(256)
void k_resid_ln(const float* __restrict__ x, const float* __restrict__ res_in,
                float* __restrict__ res_out, const float* __restrict__ w,
                const float* __restrict__ b, _Float16* __restrict__ xn_h,
                int first) {
    __shared__ float sbuf[256];
    int row = blockIdx.x;
    const float* xr = x + (size_t)row * DD;
    float v[3];
    float s = 0.f;
    #pragma unroll
    for (int i = 0; i < 3; ++i) { v[i] = xr[threadIdx.x + i * 256]; s += v[i]; }
    float mu = block_sum256(s, sbuf) * (1.f / DD);
    float q = 0.f;
    #pragma unroll
    for (int i = 0; i < 3; ++i) { float d0 = v[i] - mu; q += d0 * d0; }
    float var  = block_sum256(q, sbuf) * (1.f / DD);
    float rstd = rsqrtf(var + EPSV);
    #pragma unroll
    for (int i = 0; i < 3; ++i) {
        int col = threadIdx.x + i * 256;
        size_t o = (size_t)row * DD + col;
        xn_h[o] = (_Float16)((v[i] - mu) * rstd * w[col] + b[col]);
        res_out[o] = first ? v[i] : (v[i] + res_in[o]);
    }
}

// ============================================================
// Final LayerNorm of (x + res) -> out (f32)
// ============================================================
__global__ __launch_bounds__(256)
void k_final_ln(const float* __restrict__ x, const float* __restrict__ res,
                const float* __restrict__ w, const float* __restrict__ b,
                float* __restrict__ out) {
    __shared__ float sbuf[256];
    int row = blockIdx.x;
    float v[3];
    float s = 0.f;
    #pragma unroll
    for (int i = 0; i < 3; ++i) {
        size_t o = (size_t)row * DD + threadIdx.x + i * 256;
        v[i] = x[o] + res[o];
        s += v[i];
    }
    float mu = block_sum256(s, sbuf) * (1.f / DD);
    float q = 0.f;
    #pragma unroll
    for (int i = 0; i < 3; ++i) { float d0 = v[i] - mu; q += d0 * d0; }
    float var  = block_sum256(q, sbuf) * (1.f / DD);
    float rstd = rsqrtf(var + EPSV);
    #pragma unroll
    for (int i = 0; i < 3; ++i) {
        int col = threadIdx.x + i * 256;
        out[(size_t)row * DD + col] = (v[i] - mu) * rstd * w[col] + b[col];
    }
}

// ============================================================
// Generic cast: packed f16 dst[rowsP x colsP] from f32 src[rows x cols] (ld),
// zero padding outside.
// ============================================================
__global__ __launch_bounds__(256)
void k_cast(const float* __restrict__ src, _Float16* __restrict__ dst,
            int rows, int cols, int ld, int colsP, int total) {
    int idx = blockIdx.x * 256 + threadIdx.x;
    if (idx >= total) return;
    int r = idx / colsP, c = idx % colsP;
    float v = (r < rows && c < cols) ? src[(size_t)r * ld + c] : 0.f;
    dst[idx] = (_Float16)v;
}

// ============================================================
// WMMA GEMM:  C[M,N] = A[M,Kp] * B[Np,Kp]^T   (f16 in, f32 out)
// A, B packed f16 (lda = ldb = Kp, multiples of 32; tiles exact).
// Block tile 128x64, 8 waves as 4(m) x 2(n); wave tile 32x32 = 4 WMMAs.
// Double-buffered LDS; async global->LDS b128 staging when available.
// LDS row stride 40 halves (80B): 16B-aligned b128 writes, 16 rows
// hit 16 distinct banks for fragment reads.
// ============================================================
#define BM 128
#define BN 64
#define BK 32
#define LSTR 40
#define A_TILE_H (BM * LSTR)   // 5120 halves
#define B_TILE_H (BN * LSTR)   // 2560 halves
#define BUF_H    (A_TILE_H + B_TILE_H)

__device__ __forceinline__ void stage_tiles(const _Float16* __restrict__ Ag,
                                            const _Float16* __restrict__ Bg,
                                            _Float16* As, _Float16* Bs,
                                            int Kp, int k0, int tid) {
    // A tile: 128 rows x 32 halves = 512 x 16B chunks -> 2 per thread
    #pragma unroll
    for (int s = 0; s < 2; ++s) {
        int chunk = tid + s * 256;
        int r = chunk >> 2, sg = chunk & 3;
        async_b128(Ag + (size_t)r * Kp + k0 + sg * 8, As + r * LSTR + sg * 8);
    }
    // B tile: 64 rows x 32 halves = 256 chunks -> 1 per thread
    {
        int r = tid >> 2, sg = tid & 3;
        async_b128(Bg + (size_t)r * Kp + k0 + sg * 8, Bs + r * LSTR + sg * 8);
    }
}

// fragment gather per CDNA5 16-bit A/B layout:
// elem e=2v+h : k = (v<4 ? 2v : 16+2(v-4)) + h + 8*lgrp
__device__ __forceinline__ v16h load_frag(const _Float16* __restrict__ rowp,
                                          int lgrp) {
    v16h f = {};
    #pragma unroll
    for (int v = 0; v < 8; ++v) {
        int kk = ((v < 4) ? (v * 2) : (16 + (v - 4) * 2)) + lgrp * 8;
        f[2 * v]     = rowp[kk];
        f[2 * v + 1] = rowp[kk + 1];
    }
    return f;
}

__global__ __launch_bounds__(256)
void k_gemm(const _Float16* __restrict__ A, const _Float16* __restrict__ Bm,
            float* __restrict__ C, int Nstore, int Kp, int ldc) {
    __shared__ _Float16 sm[2 * BUF_H];

    const int tid  = threadIdx.x;
    const int wave = tid >> 5;
    const int lane = tid & 31;
    const int lgrp = lane >> 4;
    const int lrow = lane & 15;
    const int wm = (wave & 3) * 32;     // wave row offset in 128
    const int wn = (wave >> 2) * 32;    // wave col offset in 64

    const int tile_m = blockIdx.y * BM;
    const int tile_n = blockIdx.x * BN;
    const _Float16* Ag = A + (size_t)tile_m * Kp;
    const _Float16* Bg = Bm + (size_t)tile_n * Kp;

    v8f acc00 = {}, acc01 = {}, acc10 = {}, acc11 = {};

    const int nk = Kp / BK;
    stage_tiles(Ag, Bg, sm, sm + A_TILE_H, Kp, 0, tid);
    async_wait0();
    __syncthreads();

    for (int i = 0; i < nk; ++i) {
        _Float16* buf = sm + (i & 1) * BUF_H;
        if (i + 1 < nk) {
            _Float16* nxt = sm + ((i + 1) & 1) * BUF_H;
            stage_tiles(Ag, Bg, nxt, nxt + A_TILE_H, Kp, (i + 1) * BK, tid);
        }
        const _Float16* As = buf;
        const _Float16* Bs = buf + A_TILE_H;

        v16h a0 = load_frag(&As[(wm + lrow) * LSTR], lgrp);
        v16h a1 = load_frag(&As[(wm + 16 + lrow) * LSTR], lgrp);
        v16h b0 = load_frag(&Bs[(wn + lrow) * LSTR], lgrp);
        v16h b1 = load_frag(&Bs[(wn + 16 + lrow) * LSTR], lgrp);

        acc00 = __builtin_amdgcn_wmma_f32_16x16x32_f16(false, a0, false, b0,
                                                       (short)0, acc00, false, false);
        acc01 = __builtin_amdgcn_wmma_f32_16x16x32_f16(false, a0, false, b1,
                                                       (short)0, acc01, false, false);
        acc10 = __builtin_amdgcn_wmma_f32_16x16x32_f16(false, a1, false, b0,
                                                       (short)0, acc10, false, false);
        acc11 = __builtin_amdgcn_wmma_f32_16x16x32_f16(false, a1, false, b1,
                                                       (short)0, acc11, false, false);

        async_wait0();
        __syncthreads();
    }

    // store: lane n = lrow; VGPR r -> m = r + 8*lgrp within 16x16 tile
    int gn0 = tile_n + wn + lrow;
    int gn1 = gn0 + 16;
    #pragma unroll
    for (int r = 0; r < 8; ++r) {
        int gm0 = tile_m + wm + r + lgrp * 8;
        int gm1 = gm0 + 16;
        if (gn0 < Nstore) {
            C[(size_t)gm0 * ldc + gn0] = acc00[r];
            C[(size_t)gm1 * ldc + gn0] = acc10[r];
        }
        if (gn1 < Nstore) {
            C[(size_t)gm0 * ldc + gn1] = acc01[r];
            C[(size_t)gm1 * ldc + gn1] = acc11[r];
        }
    }
}

// ============================================================
// Depthwise causal conv (KC=4) + bias + SiLU.
// xi = xz[..., :DI] (row stride 2*DI). Writes f32 (scan) + f16 (GEMM).
// ============================================================
__global__ __launch_bounds__(256)
void k_conv_silu(const float* __restrict__ xz, const float* __restrict__ cw,
                 const float* __restrict__ cb, float* __restrict__ xc,
                 _Float16* __restrict__ xc_h) {
    int idx = blockIdx.x * 256 + threadIdx.x;
    if (idx >= ROWS * DI) return;
    int c = idx % DI;
    int l = (idx / DI) % LL;
    int b = idx / (DI * LL);
    float acc = cb[c];
    #pragma unroll
    for (int j = 0; j < KC; ++j) {
        int ls = l - (KC - 1) + j;
        if (ls >= 0)
            acc += xz[((size_t)(b * LL + ls)) * (2 * DI) + c] * cw[c * KC + j];
    }
    float r = acc / (1.f + __expf(-acc));   // SiLU
    xc[idx]   = r;
    xc_h[idx] = (_Float16)r;
}

// ============================================================
// dt = softplus(dtb + bias[c])  (in-place safe)
// ============================================================
__global__ __launch_bounds__(256)
void k_softplus(const float* __restrict__ dtb, const float* __restrict__ bdt,
                float* __restrict__ dt) {
    int idx = blockIdx.x * 256 + threadIdx.x;
    if (idx >= ROWS * DI) return;
    int c = idx % DI;
    float v = dtb[idx] + bdt[c];
    dt[idx] = (v > 20.f) ? v : log1pf(__expf(v));
}

// ============================================================
// Selective scan. One lane per (b, d, n); 16-lane shfl_xor reduction.
// Fuses D-skip and SiLU(z) gate; writes y as f16 (out_proj operand).
// ============================================================
__global__ __launch_bounds__(256)
void k_scan(const float* __restrict__ dt, const float* __restrict__ dbc,
            const float* __restrict__ xc, const float* __restrict__ xz,
            const float* __restrict__ A_log, const float* __restrict__ Dp,
            _Float16* __restrict__ y) {
    int gid = blockIdx.x * 256 + threadIdx.x;   // B*DI*NS = 49152 lanes
    int n   = gid & (NS - 1);
    int grp = gid >> 4;
    int b   = grp / DI;
    int d   = grp % DI;

    float Aneg  = -__expf(A_log[d * NS + n]);
    float dskip = Dp[d];
    float h = 0.f;

    for (int l = 0; l < LL; ++l) {
        int row = b * LL + l;
        float dtv = dt[(size_t)row * DI + d];          // broadcast within group
        float xv  = xc[(size_t)row * DI + d];
        float Bt  = dbc[(size_t)row * 80 + DR + n];
        float Ct  = dbc[(size_t)row * 80 + DR + NS + n];

        h = __expf(dtv * Aneg) * h + dtv * xv * Bt;
        float p = h * Ct;
        p += __shfl_xor(p, 1);
        p += __shfl_xor(p, 2);
        p += __shfl_xor(p, 4);
        p += __shfl_xor(p, 8);
        if (n == 0) {
            float zv = xz[(size_t)row * (2 * DI) + DI + d];
            float gate = zv / (1.f + __expf(-zv));     // SiLU(z)
            y[(size_t)row * DI + d] = (_Float16)((p + xv * dskip) * gate);
        }
    }
}

// ============================================================
// Host-side orchestration
// ============================================================
extern "C" void kernel_launch(void* const* d_in, const int* in_sizes, int n_in,
                              void* d_out, int out_size, void* d_ws, size_t ws_size,
                              hipStream_t stream) {
    const float* x_in    = (const float*)d_in[0];
    const float* ln_w    = (const float*)d_in[1];
    const float* ln_b    = (const float*)d_in[2];
    const float* in_proj = (const float*)d_in[3];
    const float* conv_w  = (const float*)d_in[4];
    const float* conv_b  = (const float*)d_in[5];
    const float* x_proj  = (const float*)d_in[6];
    const float* dt_w    = (const float*)d_in[7];
    const float* dt_b    = (const float*)d_in[8];
    const float* A_log   = (const float*)d_in[9];
    const float* D_skip  = (const float*)d_in[10];
    const float* out_prj = (const float*)d_in[11];
    const float* fn_w    = (const float*)d_in[12];
    const float* fn_b    = (const float*)d_in[13];
    float* out = (float*)d_out;

    // -------- workspace partition --------
    float* ws  = (float*)d_ws;
    float* res = ws;                          // ROWS*D
    float* xb  = res + (size_t)ROWS * DD;     // ROWS*D
    float* xz  = xb  + (size_t)ROWS * DD;     // ROWS*2DI
    float* xc  = xz  + (size_t)ROWS * 2 * DI; // ROWS*DI
    float* dbc = xc  + (size_t)ROWS * DI;     // ROWS*80
    float* dtb = dbc + (size_t)ROWS * 80;     // ROWS*DI
    _Float16* h16  = (_Float16*)(dtb + (size_t)ROWS * DI);
    _Float16* xn_h = h16;                       // ROWS*768
    _Float16* xc_h = xn_h + (size_t)ROWS * DD;  // ROWS*1536
    _Float16* yb_h = xc_h + (size_t)ROWS * DI;  // ROWS*1536
    _Float16* da_h = yb_h + (size_t)ROWS * DI;  // ROWS*64  (dt GEMM A)
    _Float16* Bh   = da_h + (size_t)ROWS * 64;  // max 3072*768 (weight cast)

    const int ew_blocks = (ROWS * DI) / 256; // 12288
    #define CAST_GRID(total) (((total) + 255) / 256)

    for (int l = 0; l < NLAY; ++l) {
        const float* xcur = (l == 0) ? x_in : xb;

        // residual accumulate + LN (writes xn as f16)
        k_resid_ln<<<ROWS, 256, 0, stream>>>(xcur, res, res,
                                             ln_w + (size_t)l * DD,
                                             ln_b + (size_t)l * DD, xn_h, l == 0);

        // ---- in_proj: xz[2048,3072] = xn * Wi^T ----
        k_cast<<<CAST_GRID(2 * DI * DD), 256, 0, stream>>>(
            in_proj + (size_t)l * 2 * DI * DD, Bh, 2 * DI, DD, DD, DD, 2 * DI * DD);
        {
            dim3 g((2 * DI) / BN, ROWS / BM);
            k_gemm<<<g, 256, 0, stream>>>(xn_h, Bh, xz, 2 * DI, DD, 2 * DI);
        }

        // depthwise conv + SiLU (f32 for scan, f16 for x_proj GEMM)
        k_conv_silu<<<ew_blocks, 256, 0, stream>>>(xz,
                                                   conv_w + (size_t)l * DI * KC,
                                                   conv_b + (size_t)l * DI,
                                                   xc, xc_h);

        // ---- x_proj: dbc[2048,80] = xc * Wx^T  (N padded 80->128) ----
        k_cast<<<CAST_GRID(128 * DI), 256, 0, stream>>>(
            x_proj + (size_t)l * 80 * DI, Bh, 80, DI, DI, DI, 128 * DI);
        {
            dim3 g(128 / BN, ROWS / BM);
            k_gemm<<<g, 256, 0, stream>>>(xc_h, Bh, dbc, 80, DI, 80);
        }

        // ---- dt_proj: dtb[2048,1536] = dbc[:, :48] * Wdt^T (K padded 48->64) ----
        k_cast<<<CAST_GRID(ROWS * 64), 256, 0, stream>>>(
            dbc, da_h, ROWS, DR, 80, 64, ROWS * 64);
        k_cast<<<CAST_GRID(DI * 64), 256, 0, stream>>>(
            dt_w + (size_t)l * DI * DR, Bh, DI, DR, DR, 64, DI * 64);
        {
            dim3 g(DI / BN, ROWS / BM);
            k_gemm<<<g, 256, 0, stream>>>(da_h, Bh, dtb, DI, 64, DI);
        }
        k_softplus<<<ew_blocks, 256, 0, stream>>>(dtb, dt_b + (size_t)l * DI, dtb);

        // selective scan + gate (writes y as f16)
        k_scan<<<(BB * DI * NS) / 256, 256, 0, stream>>>(
            dtb, dbc, xc, xz, A_log + (size_t)l * DI * NS,
            D_skip + (size_t)l * DI, yb_h);

        // ---- out_proj: xb[2048,768] = y * Wo^T ----
        k_cast<<<CAST_GRID(DD * DI), 256, 0, stream>>>(
            out_prj + (size_t)l * DD * DI, Bh, DD, DI, DI, DI, DD * DI);
        {
            dim3 g(DD / BN, ROWS / BM);
            k_gemm<<<g, 256, 0, stream>>>(yb_h, Bh, xb, DD, DI, DD);
        }
    }

    // final LN(x + res)
    k_final_ln<<<ROWS, 256, 0, stream>>>(xb, res, fn_w, fn_b, out);
}
